// _LyricEncoder_55095840473766
// MI455X (gfx1250) — matrix operand order repaired and checked by reading, not compiled, not run on hardware
//
#include <hip/hip_runtime.h>
#include <hip/hip_bf16.h>
#include <math.h>

// ---------------------------------------------------------------------------
// Model dimensions (fixed by the reference)
// ---------------------------------------------------------------------------
namespace {
constexpr int kB   = 2;
constexpr int kL   = 1024;
constexpr int kTXT = 1024;
constexpr int kHID = 2048;
constexpr int kINT = 6144;
constexpr int kNH  = 16;
constexpr int kNKV = 8;
constexpr int kHD  = 128;
constexpr int kNL  = 8;
constexpr int kSW  = 128;
constexpr int kM   = kB * kL;          // 2048 tokens
constexpr float kEPS = 1e-6f;
constexpr float kScale = 0.0883883476483184405f;  // 1/sqrt(128)
}

typedef __attribute__((ext_vector_type(2)))  _Float16 v2h;
typedef __attribute__((ext_vector_type(8)))  _Float16 v8h;
typedef __attribute__((ext_vector_type(16))) _Float16 v16h;
typedef __attribute__((ext_vector_type(8)))  float    v8f;

// ---------------------------------------------------------------------------
// Async global -> LDS copy (CDNA5 GLOBAL_LOAD_ASYNC_TO_LDS_B128, ASYNCcnt).
// LDS address operand: low 32 bits of the flat shared-aperture address
// (ISA 10.2).
// ---------------------------------------------------------------------------
__device__ __forceinline__ void async_ld_b128(unsigned lds_off, const float* gsrc)
{
    asm volatile("global_load_async_to_lds_b128 %0, %1, off"
                 :: "v"(lds_off), "v"(gsrc) : "memory");
}
__device__ __forceinline__ void wait_async0()
{
    asm volatile("s_wait_asynccnt 0x0" ::: "memory");
}

// ---------------------------------------------------------------------------
// WMMA tiled GEMM:  C[M,N] = A[M,K] @ B[K,N]  (+ bias[n]) (+ Res[m,n])
// fp32 in/out, f16 WMMA accumulate in fp32.  N,K are compile-time.
// Block tile 128x128, 8 waves (wave32), each wave 32x64 (2x4 WMMA tiles).
// Double-buffered LDS; one barrier per 32-K step; global loads for tile i+1
// issued before the WMMAs of tile i.
// ---------------------------------------------------------------------------
#define GEMM_LDT 40   // LDS row stride in f16 elements (80B, 16B-aligned rows)

template<int N, int K>
__global__ __launch_bounds__(256)
void gemm_wmma_f16(const float* __restrict__ A, const float* __restrict__ Bw,
                   float* __restrict__ C, const float* __restrict__ bias,
                   const float* __restrict__ Res, int M)
{
    __shared__ _Float16 sA[2][128 * GEMM_LDT];   // [m 0..127][k 0..31]
    __shared__ _Float16 sB[2][128 * GEMM_LDT];   // transposed: [n 0..127][k 0..31]

    const int tid  = threadIdx.x;
    const int lane = tid & 31;
    const int wid  = tid >> 5;
    const int wm   = wid & 3;      // wave row (0..3) -> 32 rows
    const int wn   = wid >> 2;     // wave col (0..1) -> 64 cols
    const int m0   = blockIdx.y * 128;
    const int n0   = blockIdx.x * 128;

    v8f acc[2][4];
#pragma unroll
    for (int i = 0; i < 2; ++i)
#pragma unroll
        for (int j = 0; j < 4; ++j)
#pragma unroll
            for (int g = 0; g < 8; ++g) acc[i][j][g] = 0.0f;

    // ---- staging thread mapping ----
    const int ar  = tid >> 1;
    const int ac0 = (tid & 1) * 16;
    const int bk  = (tid >> 4) * 2;      // 0,2,...,30
    const int bn0 = (tid & 15) * 8;      // 0,8,...,120

    float4 a0, a1, a2, a3;               // A row segment (16 f32)
    float4 b00, b01, b10, b11;           // B rows bk (cols 0-7) and bk+1

    auto load_tiles = [&](int k0) {
        const float4* ga = (const float4*)(A + (size_t)(m0 + ar) * K + k0 + ac0);
        a0 = ga[0]; a1 = ga[1]; a2 = ga[2]; a3 = ga[3];
        const float4* g0 = (const float4*)(Bw + (size_t)(k0 + bk)     * N + n0 + bn0);
        const float4* g1 = (const float4*)(Bw + (size_t)(k0 + bk + 1) * N + n0 + bn0);
        b00 = g0[0]; b01 = g0[1]; b10 = g1[0]; b11 = g1[1];
    };

    auto store_tiles = [&](int buf) {
        _Float16* dA = &sA[buf][ar * GEMM_LDT + ac0];
        v8h p0 = { (_Float16)a0.x, (_Float16)a0.y, (_Float16)a0.z, (_Float16)a0.w,
                   (_Float16)a1.x, (_Float16)a1.y, (_Float16)a1.z, (_Float16)a1.w };
        v8h p1 = { (_Float16)a2.x, (_Float16)a2.y, (_Float16)a2.z, (_Float16)a2.w,
                   (_Float16)a3.x, (_Float16)a3.y, (_Float16)a3.z, (_Float16)a3.w };
        *(v8h*)dA       = p0;
        *(v8h*)(dA + 8) = p1;
        _Float16* dB = &sB[buf][bn0 * GEMM_LDT + bk];
        *(v2h*)(dB + 0 * GEMM_LDT) = v2h{ (_Float16)b00.x, (_Float16)b10.x };
        *(v2h*)(dB + 1 * GEMM_LDT) = v2h{ (_Float16)b00.y, (_Float16)b10.y };
        *(v2h*)(dB + 2 * GEMM_LDT) = v2h{ (_Float16)b00.z, (_Float16)b10.z };
        *(v2h*)(dB + 3 * GEMM_LDT) = v2h{ (_Float16)b00.w, (_Float16)b10.w };
        *(v2h*)(dB + 4 * GEMM_LDT) = v2h{ (_Float16)b01.x, (_Float16)b11.x };
        *(v2h*)(dB + 5 * GEMM_LDT) = v2h{ (_Float16)b01.y, (_Float16)b11.y };
        *(v2h*)(dB + 6 * GEMM_LDT) = v2h{ (_Float16)b01.z, (_Float16)b11.z };
        *(v2h*)(dB + 7 * GEMM_LDT) = v2h{ (_Float16)b01.w, (_Float16)b11.w };
    };

    // Per-lane fragment addressing (ISA 16-bit A 16x32 / B 32x16 layouts)
    const int fr   = lane & 15;
    const int akof = (lane < 16) ? 0 : 8;    // A: lanes16-31 hold K=8..15 / 24..31
    const int bkof = (lane < 16) ? 0 : 8;    // B: lanes16-31 hold K=8..15 / 24..31

    load_tiles(0);
    store_tiles(0);
    int buf = 0;

    for (int k0 = 0; k0 < K; k0 += 32) {
        __syncthreads();
        if (k0 + 32 < K) load_tiles(k0 + 32);   // HBM latency hides under WMMAs

        v16h afrag[2], bfrag[4];
#pragma unroll
        for (int tm = 0; tm < 2; ++tm) {
            const int base = (wm * 32 + tm * 16 + fr) * GEMM_LDT + akof;
            v8h lo = *(const v8h*)&sA[buf][base];
            v8h hi = *(const v8h*)&sA[buf][base + 16];
            afrag[tm] = __builtin_shufflevector(lo, hi,
                0,1,2,3,4,5,6,7,8,9,10,11,12,13,14,15);
        }
#pragma unroll
        for (int tn = 0; tn < 4; ++tn) {
            const int base = (wn * 64 + tn * 16 + fr) * GEMM_LDT + bkof;
            v8h lo = *(const v8h*)&sB[buf][base];
            v8h hi = *(const v8h*)&sB[buf][base + 16];
            bfrag[tn] = __builtin_shufflevector(lo, hi,
                0,1,2,3,4,5,6,7,8,9,10,11,12,13,14,15);
        }

#pragma unroll
        for (int tm = 0; tm < 2; ++tm)
#pragma unroll
            for (int tn = 0; tn < 4; ++tn)
                acc[tm][tn] = __builtin_amdgcn_wmma_f32_16x16x32_f16(
                    false, afrag[tm], false, bfrag[tn],
                    (short)0, acc[tm][tn], false, false);

        if (k0 + 32 < K) store_tiles(buf ^ 1);
        buf ^= 1;
    }

    // ---- epilogue: batched residual loads, immediate-offset addressing ----
    const int crow0 = (lane < 16) ? 0 : 8;
    const int ccol  = lane & 15;
#pragma unroll
    for (int tm = 0; tm < 2; ++tm) {
#pragma unroll
        for (int tn = 0; tn < 4; ++tn) {
            const int rb = m0 + wm * 32 + tm * 16 + crow0;
            const int cb = n0 + wn * 64 + tn * 16 + ccol;
            float*       cp = C + (size_t)rb * N + cb;
            v8f a = acc[tm][tn];
            if (Res) {
                const float* rp = Res + (size_t)rb * N + cb;
                float rv[8];
#pragma unroll
                for (int g = 0; g < 8; ++g) rv[g] = rp[(size_t)g * N];
#pragma unroll
                for (int g = 0; g < 8; ++g) a[g] += rv[g];
            }
            if (bias) {
                const float bv = bias[cb];
#pragma unroll
                for (int g = 0; g < 8; ++g) a[g] += bv;
            }
#pragma unroll
            for (int g = 0; g < 8; ++g) cp[(size_t)g * N] = a[g];
        }
    }
    (void)M;
}

// ---------------------------------------------------------------------------
// Row RMSNorm: Y[row] = W * X[row] * rsqrt(mean(X^2) + eps), D % 1024 == 0
// ---------------------------------------------------------------------------
__global__ __launch_bounds__(256)
void rmsnorm_kernel(const float* __restrict__ X, const float* __restrict__ W,
                    float* __restrict__ Y, int D)
{
    __shared__ float red[256];
    const int row = blockIdx.x;
    const float* x = X + (size_t)row * D;
    float*       y = Y + (size_t)row * D;
    float s = 0.0f;
    for (int d = threadIdx.x * 4; d < D; d += 1024) {
        float4 v = *(const float4*)(x + d);
        s += v.x * v.x + v.y * v.y + v.z * v.z + v.w * v.w;
    }
    red[threadIdx.x] = s;
    __syncthreads();
    for (int off = 128; off > 0; off >>= 1) {
        if (threadIdx.x < off) red[threadIdx.x] += red[threadIdx.x + off];
        __syncthreads();
    }
    const float inv = rsqrtf(red[0] / (float)D + kEPS);
    for (int d = threadIdx.x * 4; d < D; d += 1024) {
        float4 v = *(const float4*)(x + d);
        float4 w = *(const float4*)(W + d);
        float4 o;
        o.x = w.x * v.x * inv; o.y = w.y * v.y * inv;
        o.z = w.z * v.z * inv; o.w = w.w * v.w * inv;
        *(float4*)(y + d) = o;
    }
}

// ---------------------------------------------------------------------------
// Fused per-head RMSNorm (over HD=128) + RoPE (base 1e6), in place.
// ---------------------------------------------------------------------------
__global__ __launch_bounds__(128)
void headnorm_rope(float* __restrict__ X, const float* __restrict__ W, int nheads)
{
    __shared__ float red[128];
    __shared__ float buf[128];
    int idx = blockIdx.x;
    const int hh  = idx % nheads;  idx /= nheads;
    const int pos = idx % kL;
    const int b   = idx / kL;
    float* x = X + (((size_t)b * kL + pos) * nheads + hh) * kHD;

    const int d = threadIdx.x;
    const float v = x[d];
    red[d] = v * v;
    __syncthreads();
    for (int off = 64; off > 0; off >>= 1) {
        if (d < off) red[d] += red[d + off];
        __syncthreads();
    }
    const float inv = rsqrtf(red[0] / (float)kHD + kEPS);
    const float nv  = W[d] * v * inv;
    buf[d] = nv;
    __syncthreads();

    const int   dh       = d & 63;
    const float inv_freq = __powf(1000000.0f, -((float)(2 * dh)) / 128.0f);
    const float ang      = (float)pos * inv_freq;
    float sn, cs;
    __sincosf(ang, &sn, &cs);
    const float other = (d < 64) ? -buf[d + 64] : buf[d - 64];
    x[d] = nv * cs + other * sn;
}

// ---------------------------------------------------------------------------
// Flash attention, LDS-tiled, 32-keys-in-parallel softmax, double-buffered
// async global->LDS staging (ASYNCcnt) so tile i+1's fetch overlaps tile i's
// compute.  Block: 256 threads = 8 waves, one (b, head), 8 query rows.
// ---------------------------------------------------------------------------
__global__ __launch_bounds__(256)
void flash_attn(const float* __restrict__ Q, const float* __restrict__ Kb,
                const float* __restrict__ Vb, float* __restrict__ O,
                const int* __restrict__ amask, int sliding)
{
    __shared__ float sQ[8 * 128];
    __shared__ float sK[2][32 * 128];
    __shared__ float sV[2][32 * 128];

    const int tid  = threadIdx.x;
    const int lane = tid & 31;
    const int wid  = tid >> 5;
    int idx = blockIdx.x;                       // over B * NH * (L/8)
    const int qblk = idx % (kL / 8);  idx /= (kL / 8);
    const int h    = idx % kNH;       idx /= kNH;
    const int b    = idx;
    const int qbase = qblk * 8;
    const int qpos  = qbase + wid;
    const int kvh   = h / (kNH / kNKV);

    // q row (pre-scaled) into LDS; visible after first in-loop barrier
    const size_t qoff = (((size_t)b * kL + qpos) * kNH + h) * kHD;
    for (int d = lane; d < kHD; d += 32) sQ[wid * 128 + d] = Q[qoff + d] * kScale;

    float m = -INFINITY, l = 0.0f;
    float4 o4; o4.x = o4.y = o4.z = o4.w = 0.0f;

    int klo = 0, khi = kL - 1;
    if (sliding) {
        klo = qbase - kSW;     if (klo < 0) klo = 0;
        khi = qbase + 7 + kSW; if (khi > kL - 1) khi = kL - 1;
    }

    const int srow = tid >> 3;           // staging: key row 0..31
    const int sseg = (tid & 7) * 16;     // staging: 16-float segment

    auto stage_async = [&](int t0, int buf) {
        int kst = t0 + srow; if (kst > kL - 1) kst = kL - 1;
        const size_t rowoff = (((size_t)b * kL + kst) * kNKV + kvh) * kHD + sseg;
        const float* gk = Kb + rowoff;
        const float* gv = Vb + rowoff;
        const unsigned lk = (unsigned)(size_t)&sK[buf][srow * 128 + sseg];
        const unsigned lv = (unsigned)(size_t)&sV[buf][srow * 128 + sseg];
#pragma unroll
        for (int j = 0; j < 4; ++j) {
            async_ld_b128(lk + 16 * j, gk + 4 * j);
            async_ld_b128(lv + 16 * j, gv + 4 * j);
        }
    };

    const int t0_first = klo & ~31;
    stage_async(t0_first, 0);
    int buf = 0;

    for (int t0 = t0_first; t0 <= khi; t0 += 32) {
        wait_async0();        // this wave's async copies for `buf` complete
        __syncthreads();      // all waves' copies visible; prior reads of buf^1 done
        if (t0 + 32 <= khi) stage_async(t0 + 32, buf ^ 1);   // overlap next fetch

        // ---- scores: this lane owns key (t0 + lane) ----
        const int kp = t0 + lane;
        const float4* q4 = (const float4*)&sQ[wid * 128];
        const float4* k4 = (const float4*)&sK[buf][lane * 128];
        float s = 0.0f;
#pragma unroll
        for (int t = 0; t < 32; ++t) {
            const int c = (t + lane) & 31;      // rotated: conflict-free banks
            float4 qv = q4[c];
            float4 kv = k4[c];
            s += qv.x * kv.x + qv.y * kv.y + qv.z * kv.z + qv.w * kv.w;
        }
        bool valid = (kp < kL) && (amask[b * kL + (kp < kL ? kp : kL - 1)] != 0);
        if (sliding) valid = valid && (kp >= qpos - kSW) && (kp <= qpos + kSW);
        if (!valid) s = -INFINITY;

        // ---- one online-softmax update per 32 keys ----
        float tmax = s;
#pragma unroll
        for (int off = 16; off > 0; off >>= 1)
            tmax = fmaxf(tmax, __shfl_xor(tmax, off, 32));
        const float nm = fmaxf(m, tmax);
        if (nm != -INFINITY) {                  // wave-uniform
            const float corr = __expf(m - nm);  // m==-inf -> 0
            const float p    = __expf(s - nm);  // s==-inf -> 0
            float ts = p;
#pragma unroll
            for (int off = 16; off > 0; off >>= 1) ts += __shfl_xor(ts, off, 32);
            l = l * corr + ts;
            o4.x *= corr; o4.y *= corr; o4.z *= corr; o4.w *= corr;
            const float4* v4 = (const float4*)&sV[buf][0];
#pragma unroll
            for (int kk = 0; kk < 32; ++kk) {
                const float pk = __shfl(p, kk, 32);
                float4 vv = v4[kk * 32 + lane];  // lane owns dims 4*lane..+3
                o4.x += pk * vv.x; o4.y += pk * vv.y;
                o4.z += pk * vv.z; o4.w += pk * vv.w;
            }
            m = nm;
        }
        buf ^= 1;
    }

    const float inv = 1.0f / l;
    float4 r;
    r.x = o4.x * inv; r.y = o4.y * inv; r.z = o4.z * inv; r.w = o4.w * inv;
    *(float4*)&O[qoff + lane * 4] = r;
}

// ---------------------------------------------------------------------------
// g = silu(g) * u, elementwise (vectorized by 4)
// ---------------------------------------------------------------------------
__global__ __launch_bounds__(256)
void silu_mul(float4* __restrict__ g, const float4* __restrict__ u, int n4)
{
    const int i = blockIdx.x * 256 + threadIdx.x;
    if (i < n4) {
        float4 x = g[i];
        float4 y = u[i];
        float4 o;
        o.x = (x.x / (1.0f + __expf(-x.x))) * y.x;
        o.y = (x.y / (1.0f + __expf(-x.y))) * y.y;
        o.z = (x.z / (1.0f + __expf(-x.z))) * y.z;
        o.w = (x.w / (1.0f + __expf(-x.w))) * y.w;
        g[i] = o;
    }
}

// ---------------------------------------------------------------------------
// Host orchestration
// ---------------------------------------------------------------------------
extern "C" void kernel_launch(void* const* d_in, const int* in_sizes, int n_in,
                              void* d_out, int out_size, void* d_ws, size_t ws_size,
                              hipStream_t stream)
{
    (void)in_sizes; (void)n_in; (void)out_size; (void)ws_size;

    const float* x        = (const float*)d_in[0];
    const int*   amask    = (const int*)  d_in[1];
    const float* W_embed  = (const float*)d_in[2];
    const float* b_embed  = (const float*)d_in[3];
    const float* Wq       = (const float*)d_in[4];
    const float* Wk       = (const float*)d_in[5];
    const float* Wv       = (const float*)d_in[6];
    const float* Wo       = (const float*)d_in[7];
    const float* w_in_ln  = (const float*)d_in[8];
    const float* w_post   = (const float*)d_in[9];
    const float* w_qn     = (const float*)d_in[10];
    const float* w_kn     = (const float*)d_in[11];
    const float* Wg       = (const float*)d_in[12];
    const float* Wu       = (const float*)d_in[13];
    const float* Wd       = (const float*)d_in[14];
    const float* w_final  = (const float*)d_in[15];
    float* out = (float*)d_out;

    // Workspace carve-up (floats)
    float* ws = (float*)d_ws;
    float* h    = ws;                         // [M, HID]
    float* hn   = h    + (size_t)kM * kHID;   // [M, HID]
    float* qb   = hn   + (size_t)kM * kHID;   // [M, NH*HD]
    float* kb   = qb   + (size_t)kM * kNH  * kHD;  // [M, NKV*HD]
    float* vb   = kb   + (size_t)kM * kNKV * kHD;  // [M, NKV*HD]
    float* ob   = vb   + (size_t)kM * kNKV * kHD;  // [M, NH*HD]
    float* gb   = ob   + (size_t)kM * kNH  * kHD;  // [M, INT]
    float* ub   = gb   + (size_t)kM * kINT;        // [M, INT]

    #define GEMM_LAUNCH(NN, KK, Aa, Bb, Cc, bias, res)                        \
        gemm_wmma_f16<NN, KK><<<dim3((NN)/128, kM/128), 256, 0, stream>>>(    \
            (Aa), (Bb), (Cc), (bias), (res), kM)

    // h = x @ W_embed + b_embed
    GEMM_LAUNCH(kHID, kTXT, x, W_embed, h, b_embed, (const float*)nullptr);

    for (int i = 0; i < kNL; ++i) {
        // pre-attention norm
        rmsnorm_kernel<<<kM, 256, 0, stream>>>(h, w_in_ln + (size_t)i * kHID, hn, kHID);

        // QKV projections
        GEMM_LAUNCH(2048, 2048, hn, Wq + (size_t)i * kHID * (kNH  * kHD), qb, nullptr, nullptr);
        GEMM_LAUNCH(1024, 2048, hn, Wk + (size_t)i * kHID * (kNKV * kHD), kb, nullptr, nullptr);
        GEMM_LAUNCH(1024, 2048, hn, Wv + (size_t)i * kHID * (kNKV * kHD), vb, nullptr, nullptr);

        // q/k head norm + RoPE (in place)
        headnorm_rope<<<kB * kL * kNH,  128, 0, stream>>>(qb, w_qn + (size_t)i * kHD, kNH);
        headnorm_rope<<<kB * kL * kNKV, 128, 0, stream>>>(kb, w_kn + (size_t)i * kHD, kNKV);

        // attention (even layers: sliding window, odd: full)
        flash_attn<<<kB * kNH * (kL / 8), 256, 0, stream>>>(
            qb, kb, vb, ob, amask, (i % 2 == 0) ? 1 : 0);

        // h = h + o @ Wo
        GEMM_LAUNCH(2048, 2048, ob, Wo + (size_t)i * (kNH * kHD) * kHID, h, nullptr, h);

        // post norm + MLP
        rmsnorm_kernel<<<kM, 256, 0, stream>>>(h, w_post + (size_t)i * kHID, hn, kHID);
        GEMM_LAUNCH(6144, 2048, hn, Wg + (size_t)i * kHID * kINT, gb, nullptr, nullptr);
        GEMM_LAUNCH(6144, 2048, hn, Wu + (size_t)i * kHID * kINT, ub, nullptr, nullptr);

        const int n4 = (kM * kINT) / 4;
        silu_mul<<<(n4 + 255) / 256, 256, 0, stream>>>((float4*)gb, (const float4*)ub, n4);

        // h = h + (silu(g)*u) @ Wd
        GEMM_LAUNCH(2048, 6144, gb, Wd + (size_t)i * kINT * kHID, h, nullptr, h);
    }

    // final norm -> out
    rmsnorm_kernel<<<kM, 256, 0, stream>>>(h, w_final, out, kHID);

    #undef GEMM_LAUNCH
}
